// fixation_ViT_65953517798013
// MI455X (gfx1250) — compile-verified, matrix operand order checked by reading, not run on hardware
//
#include <hip/hip_runtime.h>
#include <cstdint>
#include <cstddef>

// ---------------- model dims ----------------
#define VIT_B   32
#define VIT_N   196
#define VIT_S   197
#define VIT_D   768
#define VIT_NH  12
#define VIT_HD  64
#define VIT_L   12
#define VIT_F   3072
#define VIT_NL  1000
#define IMGSZ   224
#define TOK_M   (VIT_B * VIT_S)   // 6304
#define PAT_M   (VIT_B * VIT_N)   // 6272

typedef __bf16 bf16_t;
typedef __attribute__((ext_vector_type(16))) __bf16        v16bf;
typedef __attribute__((ext_vector_type(8)))  float         v8f;
typedef __attribute__((ext_vector_type(4)))  unsigned int  u32x4;

__device__ inline bf16_t f2bf(float f) { return (bf16_t)f; }
__device__ inline float  bf2f(bf16_t h) { return (float)h; }

union FragU { v16bf v; u32x4 q[2]; };

// Async DMA: move 16B/lane global -> LDS, tracked by ASYNCcnt (CDNA5).
__device__ inline void async_ld_b128(const bf16_t* gptr, bf16_t* lptr) {
  unsigned lds = (unsigned)(size_t)lptr;          // low 32 bits == LDS offset
  unsigned long long ga = (unsigned long long)(size_t)gptr;
  asm volatile("global_load_async_to_lds_b128 %0, %1, off"
               :: "v"(lds), "v"(ga) : "memory");
}
__device__ inline void wait_async4() {   // wait until only the 4 just-issued remain
  asm volatile("s_wait_asynccnt 0x4" ::: "memory");
}
__device__ inline void wait_async0() {
  asm volatile("s_wait_asynccnt 0x0" ::: "memory");
}

// Load a 16x32 A-frag (or 32x16 B-frag) from a K-contiguous [rows][stride] bf16 tile.
// Lane layout per CDNA5 ISA 16-bit A/B matrix tables: lane = hi*16 + r,
// chunks at K offsets {hi*8 .. hi*8+7} and {16+hi*8 .. 16+hi*8+7}.
__device__ inline v16bf frag_ld(const bf16_t* p0, int stride) {
  int lane = threadIdx.x & 31;
  int r = lane & 15, hi = lane >> 4;
  const bf16_t* p = p0 + r * stride + hi * 8;
  FragU u;
  u.q[0] = *(const u32x4*)(p);
  u.q[1] = *(const u32x4*)(p + 16);
  return u.v;
}

__device__ inline v8f wmma_bf16(v16bf a, v16bf b, v8f c) {
  return __builtin_amdgcn_wmma_f32_16x16x32_bf16(false, a, false, b, (short)0, c,
                                                 false, false);
}

// ---------------- GEMM: C[M,N] = A[M,K](bf16) * Bt[N,K](bf16)^T + epilogue ----
// Double-buffered: async tile loads for step t+1 overlap WMMA on step t.
enum { EP_BF16 = 0, EP_GELU_BF16 = 1, EP_F32 = 2, EP_F32_RES = 3 };

template <int EPI>
__global__ __launch_bounds__(256) void gemm_bf16(
    const bf16_t* __restrict__ A, const bf16_t* __restrict__ Bt,
    const float* __restrict__ bias, const float* __restrict__ res,
    bf16_t* __restrict__ outh, float* __restrict__ outf,
    int M, int N, int K)
{
  __shared__ bf16_t sA[2][128 * 40];   // 128 rows x 32 K, padded stride 40 halfs
  __shared__ bf16_t sB[2][128 * 40];

  const int t  = threadIdx.x;
  const int bm = blockIdx.y * 128, bn = blockIdx.x * 128;
  const int w  = t >> 5, wm = w >> 2, wn = w & 3;        // 2x4 wave grid
  const int lane = t & 31, hi = lane >> 4, cl = lane & 15;

  // Per-thread tile-load coordinates (2 chunks of A + 2 of B, constant per tile).
  // Rows are clamped (not predicated) so every wave issues exactly 4 async ops
  // per tile -> the "asynccnt <= 4" wait is exact. Clamped rows are never stored.
  const int r0 = t >> 2, co = (t & 3) * 8;
  const int gmA0 = min(bm + r0, M - 1);
  const int gmA1 = min(bm + r0 + 64, M - 1);
  const int gnB0 = min(bn + r0, N - 1);
  const int gnB1 = min(bn + r0 + 64, N - 1);

  const v8f vzero = {0.f, 0.f, 0.f, 0.f, 0.f, 0.f, 0.f, 0.f};
  v8f acc[4][2];
  #pragma unroll
  for (int i = 0; i < 4; ++i)
    #pragma unroll
    for (int j = 0; j < 2; ++j) acc[i][j] = vzero;

  auto issue_tile = [&](int buf, int k0) {
    async_ld_b128(A  + (size_t)gmA0 * K + k0 + co, &sA[buf][r0 * 40 + co]);
    async_ld_b128(A  + (size_t)gmA1 * K + k0 + co, &sA[buf][(r0 + 64) * 40 + co]);
    async_ld_b128(Bt + (size_t)gnB0 * K + k0 + co, &sB[buf][r0 * 40 + co]);
    async_ld_b128(Bt + (size_t)gnB1 * K + k0 + co, &sB[buf][(r0 + 64) * 40 + co]);
  };

  const int ntile = K >> 5;
  issue_tile(0, 0);
  for (int ti = 0; ti < ntile; ++ti) {
    const int buf = ti & 1;
    if (ti + 1 < ntile) {
      issue_tile(buf ^ 1, (ti + 1) * 32);
      wait_async4();           // previous batch (this tile) fully in LDS
    } else {
      wait_async0();
    }
    __syncthreads();

    v16bf af[4];
    #pragma unroll
    for (int fm = 0; fm < 4; ++fm)
      af[fm] = frag_ld(&sA[buf][(wm * 64 + fm * 16) * 40], 40);
    v16bf bfm[2];
    #pragma unroll
    for (int fn = 0; fn < 2; ++fn)
      bfm[fn] = frag_ld(&sB[buf][(wn * 32 + fn * 16) * 40], 40);

    #pragma unroll
    for (int fm = 0; fm < 4; ++fm)
      #pragma unroll
      for (int fn = 0; fn < 2; ++fn)
        acc[fm][fn] = wmma_bf16(af[fm], bfm[fn], acc[fm][fn]);
    __syncthreads();           // readers done before next async batch overwrites
  }

  // Epilogue. C-frag layout: VGPR r -> row (hi*8 + r), col = lane&15.
  #pragma unroll
  for (int fm = 0; fm < 4; ++fm) {
    #pragma unroll
    for (int fn = 0; fn < 2; ++fn) {
      int gn = bn + wn * 32 + fn * 16 + cl;
      #pragma unroll
      for (int r = 0; r < 8; ++r) {
        int gm = bm + wm * 64 + fm * 16 + hi * 8 + r;
        if (gm < M && gn < N) {
          float v = acc[fm][fn][r];
          if (bias) v += bias[gn];
          size_t idx = (size_t)gm * N + gn;
          if constexpr (EPI == EP_GELU_BF16) {
            v = 0.5f * v * (1.0f + erff(v * 0.70710678118654752f));
            outh[idx] = f2bf(v);
          } else if constexpr (EPI == EP_BF16) {
            outh[idx] = f2bf(v);
          } else if constexpr (EPI == EP_F32) {
            outf[idx] = v;
          } else {  // EP_F32_RES
            outf[idx] = res[idx] + v;
          }
        }
      }
    }
  }
}

// ---------------- Attention: one block per (batch, head) -------------------
#define ATT_SK   (208 * 72)        // K tile, stride 72 halfs
#define ATT_SVT  (64 * 224)        // V^T, stride 224 halfs (zero-padded K)
#define ATT_SP   (8 * 16 * 224)    // per-wave P scratch
#define ATT_SMEM ((ATT_SK + ATT_SVT + ATT_SP) * 2)

__global__ __launch_bounds__(256) void k_attention(
    const bf16_t* __restrict__ qkv, bf16_t* __restrict__ attn_out)
{
  extern __shared__ char smem[];
  bf16_t* sK  = (bf16_t*)smem;
  bf16_t* sVt = sK + ATT_SK;
  bf16_t* sP  = sVt + ATT_SVT;

  const int b = blockIdx.x / VIT_NH, h = blockIdx.x % VIT_NH;
  const int t = threadIdx.x, lane = t & 31, w = t >> 5;
  const int hi = lane >> 4, cl = lane & 15;
  const bf16_t* qkvb = qkv + (size_t)b * VIT_S * (3 * VIT_D);
  const u32x4 z4 = {0u, 0u, 0u, 0u};
  const v8f vzero = {0.f, 0.f, 0.f, 0.f, 0.f, 0.f, 0.f, 0.f};

  // K tile: 208 rows x 64 halfs via async DMA (rows >=197 clamped; those score
  // columns are masked to -inf before softmax anyway).
  for (int chunk = t; chunk < 208 * 8; chunk += 256) {
    int kr = chunk >> 3, c8 = chunk & 7;
    int krc = (kr < VIT_S) ? kr : (VIT_S - 1);
    async_ld_b128(qkvb + (size_t)krc * 2304 + VIT_D + h * 64 + c8 * 8,
                  sK + kr * 72 + c8 * 8);
  }
  // V^T: sVt[d][kr], zero pad kr in [197,224)
  for (int i = t; i < 64 * 224; i += 256) {
    int d = i / 224, kr = i % 224;
    bf16_t val = f2bf(0.f);
    if (kr < VIT_S) val = qkvb[(size_t)kr * 2304 + 2 * VIT_D + h * 64 + d];
    sVt[d * 224 + kr] = val;
  }
  // zero P scratch (so padded K columns contribute 0)
  for (int i = t; i < ATT_SP / 8; i += 256) ((u32x4*)sP)[i] = z4;
  wait_async0();
  __syncthreads();

  bf16_t* Pw = sP + w * 16 * 224;
  const float scale = 0.125f;  // 1/sqrt(64)

  for (int qt0 = 0; qt0 < 13; qt0 += 8) {   // 13 q-tiles over 8 waves
    int qt = qt0 + w;
    bool act = (qt < 13);                   // wave-uniform -> EXEC all ones

    if (act) {
      // Q A-frags direct from global (rows clamped; extra rows discarded later)
      v16bf qf[2];
      #pragma unroll
      for (int kk = 0; kk < 2; ++kk) {
        int row = qt * 16 + cl; if (row > VIT_S - 1) row = VIT_S - 1;
        const bf16_t* p = qkvb + (size_t)row * 2304 + h * 64 + kk * 32 + hi * 8;
        FragU u; u.q[0] = *(const u32x4*)p; u.q[1] = *(const u32x4*)(p + 16);
        qf[kk] = u.v;
      }
      // scores -> raw (scaled / masked) into Pw
      for (int nt = 0; nt < 13; ++nt) {
        v8f sacc = vzero;
        sacc = wmma_bf16(qf[0], frag_ld(sK + nt * 16 * 72 + 0,  72), sacc);
        sacc = wmma_bf16(qf[1], frag_ld(sK + nt * 16 * 72 + 32, 72), sacc);
        int col = nt * 16 + cl;
        #pragma unroll
        for (int r = 0; r < 8; ++r) {
          int rl = hi * 8 + r;
          float v = (col < VIT_S) ? sacc[r] * scale : -1e30f;
          Pw[rl * 224 + col] = f2bf(v);
        }
      }
    }
    __syncthreads();

    if (act && lane < 16) {   // softmax, one lane per q-row
      float m = -1e30f;
      for (int c = 0; c < 208; ++c) m = fmaxf(m, bf2f(Pw[lane * 224 + c]));
      float s = 0.f;
      for (int c = 0; c < 208; ++c) {
        float e = __expf(bf2f(Pw[lane * 224 + c]) - m);
        s += e;
        Pw[lane * 224 + c] = f2bf(e);
      }
      float inv = 1.0f / s;
      for (int c = 0; c < 208; ++c)
        Pw[lane * 224 + c] = f2bf(bf2f(Pw[lane * 224 + c]) * inv);
    }
    __syncthreads();

    if (act) {
      // O = P(16x224) @ V^T-layout(64x224)^T -> 4 fragments of 16 cols
      #pragma unroll
      for (int f = 0; f < 4; ++f) {
        v8f oacc = vzero;
        #pragma unroll
        for (int kk = 0; kk < 7; ++kk)
          oacc = wmma_bf16(frag_ld(Pw + kk * 32, 224),
                           frag_ld(sVt + (f * 16) * 224 + kk * 32, 224), oacc);
        #pragma unroll
        for (int r = 0; r < 8; ++r) {
          int row = qt * 16 + hi * 8 + r;
          if (row < VIT_S)
            attn_out[((size_t)(b * VIT_S + row)) * VIT_D + h * 64 + f * 16 + cl] =
                f2bf(oacc[r]);
        }
      }
    }
    __syncthreads();
  }
}

// ---------------- LayerNorm: one block per row, 768 elems -------------------
__global__ __launch_bounds__(256) void k_layernorm(
    const float* __restrict__ x, const float* __restrict__ g,
    const float* __restrict__ bta, bf16_t* __restrict__ out, int row_stride)
{
  int r = blockIdx.x, t = threadIdx.x;
  const float* xr = x + (size_t)r * row_stride * VIT_D;
  float v0 = xr[t], v1 = xr[t + 256], v2 = xr[t + 512];
  __shared__ float rs[256], rq[256];
  rs[t] = v0 + v1 + v2;
  rq[t] = v0 * v0 + v1 * v1 + v2 * v2;
  __syncthreads();
  for (int o = 128; o > 0; o >>= 1) {
    if (t < o) { rs[t] += rs[t + o]; rq[t] += rq[t + o]; }
    __syncthreads();
  }
  float mu = rs[0] * (1.f / 768.f);
  float var = rq[0] * (1.f / 768.f) - mu * mu;
  float rstd = rsqrtf(var + 1e-12f);
  bf16_t* orow = out + (size_t)r * VIT_D;
  orow[t]       = f2bf((v0 - mu) * rstd * g[t]       + bta[t]);
  orow[t + 256] = f2bf((v1 - mu) * rstd * g[t + 256] + bta[t + 256]);
  orow[t + 512] = f2bf((v2 - mu) * rstd * g[t + 512] + bta[t + 512]);
}

// ---------------- elementwise helpers ---------------------------------------
__global__ void k_transpose_bf16(const float* __restrict__ in,
                                 bf16_t* __restrict__ out, int L, int K, int Nn) {
  int i = blockIdx.x * blockDim.x + threadIdx.x;
  if (i >= L * K * Nn) return;
  int k = i % K, n = (i / K) % Nn, l = i / (K * Nn);
  out[i] = f2bf(in[((size_t)l * K + k) * Nn + n]);   // [L][K][N] -> [L][N][K]
}

__global__ void k_convert_bf16(const float* __restrict__ in,
                               bf16_t* __restrict__ out, int n) {
  int i = blockIdx.x * blockDim.x + threadIdx.x;
  if (i < n) out[i] = f2bf(in[i]);
}

__global__ void k_gather_patches(const float* __restrict__ pix,
                                 const int* __restrict__ fix,
                                 bf16_t* __restrict__ patches, int total) {
  int i = blockIdx.x * blockDim.x + threadIdx.x;
  if (i >= total) return;
  int kq = i % VIT_D, tok = i / VIT_D;
  int b = tok / VIT_N;
  int c = kq >> 8, rem = kq & 255, py = rem >> 4, px = rem & 15;
  int fx = fix[tok * 2 + 0], fy = fix[tok * 2 + 1];
  float v = 0.f;
  if (fx >= 0 && fy >= 0) {
    int top = fy - 8, left = fx - 8;
    v = pix[(((size_t)b * 3 + c) * IMGSZ + (top + py)) * IMGSZ + (left + px)];
  }
  patches[i] = f2bf(v);
}

__global__ void k_assemble(const float* __restrict__ tokens,
                           const float* __restrict__ cls_tok,
                           const float* __restrict__ pos,
                           float* __restrict__ x, int total) {
  int i = blockIdx.x * blockDim.x + threadIdx.x;
  if (i >= total) return;
  int d = i % VIT_D, s = (i / VIT_D) % VIT_S, b = i / (VIT_D * VIT_S);
  float pe = pos[(size_t)s * VIT_D + d];
  float v = (s == 0) ? (cls_tok[d] + pe)
                     : (tokens[((size_t)(b * VIT_N + (s - 1))) * VIT_D + d] + pe);
  x[i] = v;
}

// ---------------- host driver ------------------------------------------------
static inline int cdiv(int a, int b) { return (a + b - 1) / b; }

extern "C" void kernel_launch(void* const* d_in, const int* in_sizes, int n_in,
                              void* d_out, int out_size, void* d_ws, size_t ws_size,
                              hipStream_t stream) {
  (void)in_sizes; (void)n_in; (void)out_size; (void)ws_size;
  const float* pixel   = (const float*)d_in[0];
  const int*   fix     = (const int*)d_in[1];
  const float* patch_w = (const float*)d_in[2];
  const float* patch_b = (const float*)d_in[3];
  const float* cls_tok = (const float*)d_in[4];
  const float* pos_emb = (const float*)d_in[5];
  const float* ln1_g   = (const float*)d_in[6];
  const float* ln1_b   = (const float*)d_in[7];
  const float* wqkv    = (const float*)d_in[8];
  const float* bqkv    = (const float*)d_in[9];
  const float* wo      = (const float*)d_in[10];
  const float* bo      = (const float*)d_in[11];
  const float* ln2_g   = (const float*)d_in[12];
  const float* ln2_b   = (const float*)d_in[13];
  const float* w1      = (const float*)d_in[14];
  const float* b1      = (const float*)d_in[15];
  const float* w2      = (const float*)d_in[16];
  const float* b2      = (const float*)d_in[17];
  const float* lnf_g   = (const float*)d_in[18];
  const float* lnf_b   = (const float*)d_in[19];
  const float* cls_w   = (const float*)d_in[20];
  const float* cls_b   = (const float*)d_in[21];

  char* p = (char*)d_ws;
  auto carve = [&](size_t bytes) {
    char* r = p; p += (bytes + 255) & ~(size_t)255; return r;
  };
  bf16_t* wqkvT  = (bf16_t*)carve((size_t)VIT_L * 3 * VIT_D * VIT_D * 2);
  bf16_t* woT    = (bf16_t*)carve((size_t)VIT_L * VIT_D * VIT_D * 2);
  bf16_t* w1T    = (bf16_t*)carve((size_t)VIT_L * VIT_F * VIT_D * 2);
  bf16_t* w2T    = (bf16_t*)carve((size_t)VIT_L * VIT_D * VIT_F * 2);
  bf16_t* pwT    = (bf16_t*)carve((size_t)VIT_D * VIT_D * 2);
  bf16_t* cwT    = (bf16_t*)carve((size_t)VIT_NL * VIT_D * 2);
  bf16_t* patches= (bf16_t*)carve((size_t)PAT_M * VIT_D * 2);
  float*  tokens = (float*) carve((size_t)PAT_M * VIT_D * 4);
  float*  x      = (float*) carve((size_t)TOK_M * VIT_D * 4);
  bf16_t* hbuf   = (bf16_t*)carve((size_t)TOK_M * VIT_D * 2);
  bf16_t* qkvbuf = (bf16_t*)carve((size_t)TOK_M * 3 * VIT_D * 2);
  bf16_t* attno  = (bf16_t*)carve((size_t)TOK_M * VIT_D * 2);
  bf16_t* ubuf   = (bf16_t*)carve((size_t)TOK_M * VIT_F * 2);
  bf16_t* hcls   = (bf16_t*)carve((size_t)VIT_B * VIT_D * 2);

  dim3 blk(256);

  // --- weight prep (fp32 -> bf16, transposed to [N][K]) ---
  { int n = VIT_L * VIT_D * 3 * VIT_D;
    k_transpose_bf16<<<cdiv(n, 256), blk, 0, stream>>>(wqkv, wqkvT, VIT_L, VIT_D, 3 * VIT_D); }
  { int n = VIT_L * VIT_D * VIT_D;
    k_transpose_bf16<<<cdiv(n, 256), blk, 0, stream>>>(wo, woT, VIT_L, VIT_D, VIT_D); }
  { int n = VIT_L * VIT_D * VIT_F;
    k_transpose_bf16<<<cdiv(n, 256), blk, 0, stream>>>(w1, w1T, VIT_L, VIT_D, VIT_F); }
  { int n = VIT_L * VIT_F * VIT_D;
    k_transpose_bf16<<<cdiv(n, 256), blk, 0, stream>>>(w2, w2T, VIT_L, VIT_F, VIT_D); }
  { int n = VIT_D * VIT_D;
    k_convert_bf16<<<cdiv(n, 256), blk, 0, stream>>>(patch_w, pwT, n); }
  { int n = VIT_NL * VIT_D;
    k_convert_bf16<<<cdiv(n, 256), blk, 0, stream>>>(cls_w, cwT, n); }

  // --- patch gather + embedding ---
  { int n = PAT_M * VIT_D;
    k_gather_patches<<<cdiv(n, 256), blk, 0, stream>>>(pixel, fix, patches, n); }
  { dim3 g(cdiv(VIT_D, 128), cdiv(PAT_M, 128));
    gemm_bf16<EP_F32><<<g, blk, 0, stream>>>(patches, pwT, patch_b, nullptr,
                                             nullptr, tokens, PAT_M, VIT_D, VIT_D); }
  { int n = TOK_M * VIT_D;
    k_assemble<<<cdiv(n, 256), blk, 0, stream>>>(tokens, cls_tok, pos_emb, x, n); }

  // --- transformer layers ---
  for (int l = 0; l < VIT_L; ++l) {
    k_layernorm<<<TOK_M, blk, 0, stream>>>(x, ln1_g + (size_t)l * VIT_D,
                                           ln1_b + (size_t)l * VIT_D, hbuf, 1);
    { dim3 g(cdiv(3 * VIT_D, 128), cdiv(TOK_M, 128));
      gemm_bf16<EP_BF16><<<g, blk, 0, stream>>>(
          hbuf, wqkvT + (size_t)l * 3 * VIT_D * VIT_D,
          bqkv + (size_t)l * 3 * VIT_D, nullptr, qkvbuf, nullptr,
          TOK_M, 3 * VIT_D, VIT_D); }
    k_attention<<<VIT_B * VIT_NH, blk, ATT_SMEM, stream>>>(qkvbuf, attno);
    { dim3 g(cdiv(VIT_D, 128), cdiv(TOK_M, 128));
      gemm_bf16<EP_F32_RES><<<g, blk, 0, stream>>>(
          attno, woT + (size_t)l * VIT_D * VIT_D,
          bo + (size_t)l * VIT_D, x, nullptr, x, TOK_M, VIT_D, VIT_D); }
    k_layernorm<<<TOK_M, blk, 0, stream>>>(x, ln2_g + (size_t)l * VIT_D,
                                           ln2_b + (size_t)l * VIT_D, hbuf, 1);
    { dim3 g(cdiv(VIT_F, 128), cdiv(TOK_M, 128));
      gemm_bf16<EP_GELU_BF16><<<g, blk, 0, stream>>>(
          hbuf, w1T + (size_t)l * VIT_F * VIT_D,
          b1 + (size_t)l * VIT_F, nullptr, ubuf, nullptr,
          TOK_M, VIT_F, VIT_D); }
    { dim3 g(cdiv(VIT_D, 128), cdiv(TOK_M, 128));
      gemm_bf16<EP_F32_RES><<<g, blk, 0, stream>>>(
          ubuf, w2T + (size_t)l * VIT_D * VIT_F,
          b2 + (size_t)l * VIT_D, x, nullptr, x, TOK_M, VIT_D, VIT_F); }
  }

  // --- final LN on CLS rows + classifier ---
  k_layernorm<<<VIT_B, blk, 0, stream>>>(x, lnf_g, lnf_b, hcls, VIT_S);
  { dim3 g(cdiv(VIT_NL, 128), cdiv(VIT_B, 128));
    gemm_bf16<EP_F32><<<g, blk, 0, stream>>>(hcls, cwT, cls_b, nullptr,
                                             nullptr, (float*)d_out,
                                             VIT_B, VIT_NL, VIT_D); }
}
